// GPT_28183575396615
// MI455X (gfx1250) — matrix operand order repaired and checked by reading, not compiled, not run on hardware
//
#include <hip/hip_runtime.h>
#include <hip/hip_bf16.h>
#include <math.h>
#include <stdint.h>

// ---------------- problem constants (match reference) ----------------
constexpr int B  = 4;
constexpr int T  = 512;
constexpr int C  = 768;
constexpr int H  = 12;
constexpr int L  = 8;
constexpr int V  = 512;
constexpr int FF = 4 * C;       // 3072
constexpr int HD = C / H;       // 64
constexpr int BT = B * T;       // 2048
constexpr long BHTT = (long)B * H * T * T;

// LDS tile strides (elements). 40 bf16 = 80 bytes: keeps 16B alignment, skews banks.
constexpr int LDS_A = 40;
constexpr int LDS_B = 40;

typedef __attribute__((ext_vector_type(16))) __bf16 v16bf;
typedef __attribute__((ext_vector_type(8)))  __bf16 v8bf;
typedef __attribute__((ext_vector_type(8)))  float  v8f;
typedef __attribute__((ext_vector_type(4)))  unsigned int v4u;
typedef __attribute__((ext_vector_type(8)))  int v8i;
typedef __attribute__((ext_vector_type(4)))  int v4i;

// ---------------- WMMA micro-kernel helpers ----------------

__device__ inline v16bf combine16(v8bf lo, v8bf hi) {
  v16bf r;
#pragma unroll
  for (int i = 0; i < 8; ++i) { r[i] = lo[i]; r[i + 8] = hi[i]; }
  return r;
}

__device__ inline v8f wmma_bf16(v16bf a, v16bf b, v8f c) {
  return __builtin_amdgcn_wmma_f32_16x16x32_bf16(
      false, a, false, b, (short)0, c, false, false);
}

// A fragment, 16-bit A 16x32 layout (ISA 7.12.2)
__device__ inline v16bf frag_a(const __bf16* sA, int mbase, int lane) {
  int r  = lane & 15;
  int kq = lane >> 4;
  const __bf16* p = sA + (mbase + r) * LDS_A;
  v8bf lo = *(const v8bf*)(p + kq * 8);        // K 0..7   or 8..15
  v8bf hi = *(const v8bf*)(p + 16 + kq * 8);   // K 16..23 or 24..31
  return combine16(lo, hi);
}

// B fragment, 16-bit B 32x16: lane holds column N=lane&15, K 8h..8h+7 then 16+8h..16+8h+7.
__device__ inline v16bf frag_b(const __bf16* sB, int nbase, int lane) {
  int c  = lane & 15;
  int kh = lane >> 4;
  const __bf16* p = sB + (nbase + c) * LDS_B;
  v8bf lo = *(const v8bf*)(p + kh * 8);
  v8bf hi = *(const v8bf*)(p + 16 + kh * 8);
  return combine16(lo, hi);
}

__device__ inline void mma_step(const __bf16* sA, const __bf16* sB,
                                int wm, int wn, int lane, v8f acc[2][2]) {
  v16bf a0 = frag_a(sA, wm,      lane);
  v16bf a1 = frag_a(sA, wm + 16, lane);
  v16bf b0 = frag_b(sB, wn,      lane);
  v16bf b1 = frag_b(sB, wn + 16, lane);
  acc[0][0] = wmma_bf16(a0, b0, acc[0][0]);
  acc[0][1] = wmma_bf16(a0, b1, acc[0][1]);
  acc[1][0] = wmma_bf16(a1, b0, acc[1][0]);
  acc[1][1] = wmma_bf16(a1, b1, acc[1][1]);
}

// Stage A tile: 128 rows x 32 K of f32 -> bf16 LDS (used by attention kernels).
__device__ inline void stage_A(__bf16* sA, const float* Abase, long lda,
                               int k0, int t) {
  int r  = t >> 1;
  int cc = (t & 1) * 16;
  const float* src = Abase + (long)r * lda + k0 + cc;
#pragma unroll
  for (int i = 0; i < 16; ++i) sA[r * LDS_A + cc + i] = (__bf16)src[i];
}

// Stage B tile when source is f32 [N][K] row-major (K contiguous).
__device__ inline void stage_B_rowK(__bf16* sB, const float* Wbase, long ldw,
                                    int k0, int t) {
  int n  = t >> 2;
  int kk = (t & 3) * 8;
  const float* src = Wbase + (long)n * ldw + k0 + kk;
#pragma unroll
  for (int i = 0; i < 8; ++i) sB[n * LDS_B + kk + i] = (__bf16)src[i];
}

// Stage B tile when source is f32 [K][N] row-major: transpose into [n][k].
__device__ inline void stage_B_colK(__bf16* sB, const float* Wbase, long ldw,
                                    int k0, int t) {
  int n  = t & 63;
  int kk = (t >> 6) * 8;
#pragma unroll
  for (int i = 0; i < 8; ++i)
    sB[n * LDS_B + kk + i] = (__bf16)Wbase[(long)(k0 + kk + i) * ldw + n];
}

// ---------------- Tensor Data Mover: 64x32 bf16 weight tile -> padded LDS ------
// D# per ISA 8.3/8.4: 2D tile, data_size=2B, tile_dim0=32 elems (64B rows),
// tile_dim1=64 rows, row stride = K elems; LDS padding 4 DWORDs every 16 DWORDs
// reproduces the 80B bank-skewed row stride of sB.
__device__ inline void tdm_load_B(const __bf16* Wt, unsigned K, unsigned Ntot,
                                  int n0, int k0, unsigned lds_off) {
  unsigned long long ga =
      (unsigned long long)(uintptr_t)(Wt + (size_t)n0 * K + (size_t)k0);
  v4u g0;
  g0[0] = 1u;                                   // count=1 (valid user D#)
  g0[1] = lds_off;                              // lds_addr (bytes)
  g0[2] = (unsigned)ga;                         // global_addr[31:0]
  g0[3] = (unsigned)(ga >> 32) | 0x80000000u;   // global_addr[56:32] | type=2
  v8i g1;
  g1[0] = (int)((1u << 16)      // data_size = 1 -> 2 bytes
              | (1u << 20)      // pad_enable
              | (3u << 22)      // pad_interval: 16 DWORDs (=64B tile row)
              | (3u << 25));    // pad_amount: 4 DWORDs (=16B pad)
  g1[1] = (int)((K & 0xFFFFu) << 16);                               // dim0 lo
  g1[2] = (int)(((K >> 16) & 0xFFFFu) | ((Ntot & 0xFFFFu) << 16));  // dim0 hi | dim1 lo
  g1[3] = (int)(((Ntot >> 16) & 0xFFFFu) | (32u << 16));            // dim1 hi | tile_dim0
  g1[4] = (int)(64u);                                               // tile_dim1 | tile_dim2=0
  g1[5] = (int)K;                                                   // dim0_stride[31:0]
  g1[6] = 0;
  g1[7] = 0;
  v4i z4 = {0, 0, 0, 0};
  v8i z8 = {0, 0, 0, 0, 0, 0, 0, 0};
  // 6-arg form (clang-23 / therock toolchain): groups 0..3, extra group, cpol
  __builtin_amdgcn_tensor_load_to_lds(g0, g1, z4, z4, z8, 0);
}

// ---------------- generic GEMM: Out[M,N] = epi(A[M,K] @ Wt^T + bias) ----------
// A: f32 [M][K]. Wt: bf16 [Ntot][K] (pre-transposed/converted weights).
// B tiles arrive via TDM; A tiles convert f32->bf16 in-register. Double-buffered.
__global__ __launch_bounds__(256) void wgemm(
    const float* __restrict__ A, int lda,
    const __bf16* __restrict__ Wt, int K, int Ntot,
    const float* __restrict__ bias,
    const float* __restrict__ residual,
    float* __restrict__ Out, int ldo,
    int fuse_gelu) {
  __shared__ __align__(16) __bf16 sA[2][128 * LDS_A];
  __shared__ __align__(16) __bf16 sB[2][64 * LDS_B];
  int t = threadIdx.x, lane = t & 31, wave = t >> 5;
  int m0 = blockIdx.y * 128, n0 = blockIdx.x * 64;
  const float* Abase = A + (long)m0 * lda;
  int wm = (wave & 3) * 32, wn = (wave >> 2) * 32;
  v8f acc[2][2] = {};

  int r = t >> 1, cc = (t & 1) * 16;
  float4 ar[4];
  auto loadA = [&](int k0) {
    const float* src = Abase + (long)r * lda + k0 + cc;
    ar[0] = *(const float4*)(src + 0);
    ar[1] = *(const float4*)(src + 4);
    ar[2] = *(const float4*)(src + 8);
    ar[3] = *(const float4*)(src + 12);
  };
  auto storeA = [&](int buf) {
    __bf16* d = &sA[buf][r * LDS_A + cc];
    const float* f = (const float*)ar;
#pragma unroll
    for (int i = 0; i < 16; ++i) d[i] = (__bf16)f[i];
  };
  bool tdm_wave = (wave == 0);

  // prologue: fill buffer 0
  loadA(0);
  if (tdm_wave) {
    tdm_load_B(Wt, (unsigned)K, (unsigned)Ntot, n0, 0,
               (unsigned)(uintptr_t)&sB[0][0]);
  }
  storeA(0);
  if (tdm_wave) __builtin_amdgcn_s_wait_tensorcnt(0);
  __syncthreads();

  int nsteps = K / 32;
  for (int s = 0; s < nsteps; ++s) {
    int cur = s & 1;
    bool more = (s + 1 < nsteps);
    if (more) {            // prefetch next tile while current computes
      loadA((s + 1) * 32);
      if (tdm_wave) {
        tdm_load_B(Wt, (unsigned)K, (unsigned)Ntot, n0, (s + 1) * 32,
                   (unsigned)(uintptr_t)&sB[cur ^ 1][0]);
      }
    }
    mma_step(sA[cur], sB[cur], wm, wn, lane, acc);
    if (more) {
      storeA(cur ^ 1);
      if (tdm_wave) __builtin_amdgcn_s_wait_tensorcnt(0);
    }
    __syncthreads();
  }

#pragma unroll
  for (int mi = 0; mi < 2; ++mi)
#pragma unroll
    for (int ni = 0; ni < 2; ++ni)
#pragma unroll
      for (int j = 0; j < 8; ++j) {
        int row = m0 + wm + mi * 16 + j + ((lane >> 4) << 3);
        int col = n0 + wn + ni * 16 + (lane & 15);
        float vv = acc[mi][ni][j];
        if (bias) vv += bias[col];
        if (fuse_gelu) vv = 0.5f * vv * (1.0f + erff(vv * 0.70710678118f));
        if (residual) vv += residual[(long)row * ldo + col];
        Out[(long)row * ldo + col] = vv;
      }
}

// ---------------- attention: scores = mask(Q K^T * scale) ----------------
__global__ __launch_bounds__(256) void attn_qk(
    const float* __restrict__ q, const float* __restrict__ kbuf,
    float* __restrict__ att) {
  int z  = blockIdx.z;           // b*H + h
  int b  = z / H, h = z % H;
  int m0 = blockIdx.y * 128;     // query tile
  int n0 = blockIdx.x * 64;      // key tile
  float* attz = att + (long)z * T * T;
  int t = threadIdx.x;
  if (n0 > m0 + 127) {           // tile entirely above diagonal
    for (int i = t; i < 128 * 64; i += 256) {
      int r = i >> 6, c = i & 63;
      attz[(long)(m0 + r) * T + n0 + c] = -1e30f;
    }
    return;
  }
  __shared__ __align__(16) __bf16 sA[128 * LDS_A];
  __shared__ __align__(16) __bf16 sB[64 * LDS_B];
  const float* Qb = q    + ((long)b * T + m0) * C + h * HD;
  const float* Kb = kbuf + ((long)b * T + n0) * C + h * HD;
  int lane = t & 31, wave = t >> 5;
  int wm = (wave & 3) * 32, wn = (wave >> 2) * 32;
  v8f acc[2][2] = {};
  for (int k0 = 0; k0 < HD; k0 += 32) {
    stage_A(sA, Qb, C, k0, t);
    stage_B_rowK(sB, Kb, C, k0, t);
    __syncthreads();
    mma_step(sA, sB, wm, wn, lane, acc);
    __syncthreads();
  }
  const float scale = 0.125f;  // 1/sqrt(64)
#pragma unroll
  for (int mi = 0; mi < 2; ++mi)
#pragma unroll
    for (int ni = 0; ni < 2; ++ni)
#pragma unroll
      for (int j = 0; j < 8; ++j) {
        int qi = m0 + wm + mi * 16 + j + ((lane >> 4) << 3);
        int kj = n0 + wn + ni * 16 + (lane & 15);
        float vv = acc[mi][ni][j] * scale;
        attz[(long)qi * T + kj] = (kj <= qi) ? vv : -1e30f;
      }
}

// ---------------- softmax rows, in place (d_out attn slab) ----------------
__global__ __launch_bounds__(256) void softmax_rows(float* __restrict__ att) {
  __shared__ float red[256];
  long row = blockIdx.x;
  float* r = att + row * (long)T;
  int t = threadIdx.x;
  float m = -3.0e38f;
  for (int c = t; c < T; c += 256) m = fmaxf(m, r[c]);
  red[t] = m; __syncthreads();
  for (int s = 128; s > 0; s >>= 1) {
    if (t < s) red[t] = fmaxf(red[t], red[t + s]);
    __syncthreads();
  }
  float mx = red[0]; __syncthreads();
  float sum = 0.f;
  for (int c = t; c < T; c += 256) { float e = __expf(r[c] - mx); r[c] = e; sum += e; }
  red[t] = sum; __syncthreads();
  for (int s = 128; s > 0; s >>= 1) {
    if (t < s) red[t] += red[t + s];
    __syncthreads();
  }
  float inv = 1.0f / red[0];
  for (int c = t; c < T; c += 256) r[c] *= inv;
}

// ---------------- y = att @ V, written directly into [B,T,C] layout ----------
__global__ __launch_bounds__(256) void attn_av(
    const float* __restrict__ att, const float* __restrict__ vbuf,
    float* __restrict__ y) {
  int z  = blockIdx.z; int b = z / H, h = z % H;
  int m0 = blockIdx.y * 128;
  __shared__ __align__(16) __bf16 sA[128 * LDS_A];
  __shared__ __align__(16) __bf16 sB[64 * LDS_B];
  const float* Ab = att + (long)z * T * T + (long)m0 * T;
  const float* Vb = vbuf + (long)b * T * C + h * HD;   // Bt[n][k] = V[k*C + n]
  int t = threadIdx.x;
  int lane = t & 31, wave = t >> 5;
  int wm = (wave & 3) * 32, wn = (wave >> 2) * 32;
  v8f acc[2][2] = {};
  for (int k0 = 0; k0 < T; k0 += 32) {
    stage_A(sA, Ab, T, k0, t);
    stage_B_colK(sB, Vb, C, k0, t);
    __syncthreads();
    mma_step(sA, sB, wm, wn, lane, acc);
    __syncthreads();
  }
#pragma unroll
  for (int mi = 0; mi < 2; ++mi)
#pragma unroll
    for (int ni = 0; ni < 2; ++ni)
#pragma unroll
      for (int j = 0; j < 8; ++j) {
        int row = m0 + wm + mi * 16 + j + ((lane >> 4) << 3);
        int col = wn + ni * 16 + (lane & 15);
        y[((long)b * T + row) * C + h * HD + col] = acc[mi][ni][j];
      }
}

// ---------------- weight pre-conversion kernels ----------------
// W [K][N] f32 -> Wt [N][K] bf16 (transpose via LDS tile)
__global__ __launch_bounds__(256) void convert_wT(const float* __restrict__ W,
                                                  __bf16* __restrict__ Wt,
                                                  int Kd, int Nd) {
  __shared__ float tile[32][33];
  int k0 = blockIdx.x * 32, n0 = blockIdx.y * 32;
  int tx = threadIdx.x & 31, ty = threadIdx.x >> 5;
  for (int i = ty; i < 32; i += 8)
    tile[i][tx] = W[(long)(k0 + i) * Nd + (n0 + tx)];
  __syncthreads();
  for (int i = ty; i < 32; i += 8)
    Wt[(long)(n0 + i) * Kd + (k0 + tx)] = (__bf16)tile[tx][i];
}

// plain f32 -> bf16 convert (already [N][K] sources, e.g. head_w)
__global__ __launch_bounds__(256) void convert_nt(const float* __restrict__ W,
                                                  __bf16* __restrict__ Wt) {
  long i0 = ((long)blockIdx.x * 256 + threadIdx.x) * 4;
#pragma unroll
  for (int j = 0; j < 4; ++j) Wt[i0 + j] = (__bf16)W[i0 + j];
}

// ---------------- elementwise kernels ----------------
__global__ void embed_kernel(const int* __restrict__ idx,
                             const float* __restrict__ tok,
                             const float* __restrict__ type_emb,
                             const float* __restrict__ pos,
                             float* __restrict__ x) {
  long n  = blockIdx.x;
  int  ti = (int)(n % T);
  int  row = idx[n];
  for (int c = threadIdx.x; c < C; c += blockDim.x)
    x[n * C + c] = tok[(long)row * C + c] + pos[(long)ti * C + c] + type_emb[C + c];
}

__global__ __launch_bounds__(256) void ln_rows(const float* __restrict__ x,
                                               const float* __restrict__ w,
                                               const float* __restrict__ bb,
                                               float* __restrict__ o) {
  __shared__ float r1[256], r2[256];
  long row = blockIdx.x;
  const float* xr = x + row * C;
  int t = threadIdx.x;
  float s = 0.f, s2 = 0.f;
  for (int c = t; c < C; c += 256) { float v = xr[c]; s += v; s2 += v * v; }
  r1[t] = s; r2[t] = s2; __syncthreads();
  for (int st = 128; st > 0; st >>= 1) {
    if (t < st) { r1[t] += r1[t + st]; r2[t] += r2[t + st]; }
    __syncthreads();
  }
  float mean = r1[0] * (1.0f / C);
  float var  = r2[0] * (1.0f / C) - mean * mean;
  float rstd = rsqrtf(var + 1e-5f);
  float* orow = o + row * C;
  for (int c = t; c < C; c += 256) orow[c] = (xr[c] - mean) * rstd * w[c] + bb[c];
}

// ---------------- host-side orchestration ----------------
extern "C" void kernel_launch(void* const* d_in, const int* in_sizes, int n_in,
                              void* d_out, int out_size, void* d_ws, size_t ws_size,
                              hipStream_t stream) {
  (void)in_sizes; (void)n_in; (void)out_size; (void)ws_size;
  const int*   idx   = (const int*)  d_in[0];
  const float* tok   = (const float*)d_in[1];
  const float* typ   = (const float*)d_in[2];
  const float* pos   = (const float*)d_in[3];
  const float* Wq    = (const float*)d_in[4];
  const float* bq    = (const float*)d_in[5];
  const float* Wk    = (const float*)d_in[6];
  const float* bk    = (const float*)d_in[7];
  const float* Wv    = (const float*)d_in[8];
  const float* bv    = (const float*)d_in[9];
  const float* Wp    = (const float*)d_in[10];
  const float* bp    = (const float*)d_in[11];
  const float* ln1w  = (const float*)d_in[12];
  const float* ln1b  = (const float*)d_in[13];
  const float* ln2w  = (const float*)d_in[14];
  const float* ln2b  = (const float*)d_in[15];
  const float* W1    = (const float*)d_in[16];
  const float* b1    = (const float*)d_in[17];
  const float* W2    = (const float*)d_in[18];
  const float* b2    = (const float*)d_in[19];
  const float* lnfw  = (const float*)d_in[20];
  const float* lnfb  = (const float*)d_in[21];
  const float* headw = (const float*)d_in[22];

  float* logits = (float*)d_out;                     // [BT, V]
  float* xout   = logits + (long)BT * V;             // [BT, C]
  float* attn   = xout + (long)BT * C;               // [L, B, H, T, T]

  float* ws = (float*)d_ws;
  const long BTC = (long)BT * C;
  float* x  = ws;              // [BT, C]
  float* h  = x  + BTC;        // [BT, C]
  float* q  = h  + BTC;        // [BT, C]
  float* k  = q  + BTC;        // [BT, C]
  float* v  = k  + BTC;        // [BT, C]
  float* y  = v  + BTC;        // [BT, C]
  float* ff = y  + BTC;        // [BT, FF]

  // bf16 pre-transposed weights, carved after f32 scratch
  __bf16* wqT = (__bf16*)(ff + (long)BT * FF);       // [L][C][C]
  __bf16* wkT = wqT + (long)L * C * C;
  __bf16* wvT = wkT + (long)L * C * C;
  __bf16* wpT = wvT + (long)L * C * C;
  __bf16* w1T = wpT + (long)L * C * C;               // [L][FF][C]
  __bf16* w2T = w1T + (long)L * C * FF;              // [L][C][FF]
  __bf16* hdT = w2T + (long)L * FF * C;              // [V][C]

  // ---- one-time (per launch) weight conversion/transposition ----
  dim3 gcc(C / 32, C / 32);      // (24,24)
  dim3 gw1(C / 32, FF / 32);     // (24,96)
  dim3 gw2(FF / 32, C / 32);     // (96,24)
  for (int l = 0; l < L; ++l) {
    convert_wT<<<gcc, 256, 0, stream>>>(Wq + (long)l * C * C,  wqT + (long)l * C * C,  C,  C);
    convert_wT<<<gcc, 256, 0, stream>>>(Wk + (long)l * C * C,  wkT + (long)l * C * C,  C,  C);
    convert_wT<<<gcc, 256, 0, stream>>>(Wv + (long)l * C * C,  wvT + (long)l * C * C,  C,  C);
    convert_wT<<<gcc, 256, 0, stream>>>(Wp + (long)l * C * C,  wpT + (long)l * C * C,  C,  C);
    convert_wT<<<gw1, 256, 0, stream>>>(W1 + (long)l * C * FF, w1T + (long)l * C * FF, C,  FF);
    convert_wT<<<gw2, 256, 0, stream>>>(W2 + (long)l * FF * C, w2T + (long)l * FF * C, FF, C);
  }
  convert_nt<<<(V * C) / 1024, 256, 0, stream>>>(headw, hdT);

  embed_kernel<<<BT, 256, 0, stream>>>(idx, tok, typ, pos, x);

  dim3 g_cc(C / 64, BT / 128);
  dim3 g_ff1(FF / 64, BT / 128);
  dim3 g_qk(T / 64, T / 128, B * H);
  dim3 g_av(1, T / 128, B * H);

  for (int l = 0; l < L; ++l) {
    const __bf16* wqT_l = wqT + (long)l * C * C;
    const __bf16* wkT_l = wkT + (long)l * C * C;
    const __bf16* wvT_l = wvT + (long)l * C * C;
    const __bf16* wpT_l = wpT + (long)l * C * C;
    const __bf16* w1T_l = w1T + (long)l * C * FF;
    const __bf16* w2T_l = w2T + (long)l * FF * C;
    float* att_l = attn + (long)l * BHTT;

    // h = ln1(x)
    ln_rows<<<BT, 256, 0, stream>>>(x, ln1w + (long)l * C, ln1b + (long)l * C, h);
    // q,k,v = h @ W + b
    wgemm<<<g_cc, 256, 0, stream>>>(h, C, wqT_l, C, C, bq + (long)l * C, nullptr, q, C, 0);
    wgemm<<<g_cc, 256, 0, stream>>>(h, C, wkT_l, C, C, bk + (long)l * C, nullptr, k, C, 0);
    wgemm<<<g_cc, 256, 0, stream>>>(h, C, wvT_l, C, C, bv + (long)l * C, nullptr, v, C, 0);
    // attention
    attn_qk<<<g_qk, 256, 0, stream>>>(q, k, att_l);
    softmax_rows<<<B * H * T, 256, 0, stream>>>(att_l);
    attn_av<<<g_av, 256, 0, stream>>>(att_l, v, y);
    // x = x + y @ Wp + bp
    wgemm<<<g_cc, 256, 0, stream>>>(y, C, wpT_l, C, C, bp + (long)l * C, x, x, C, 0);
    // h = ln2(x)
    ln_rows<<<BT, 256, 0, stream>>>(x, ln2w + (long)l * C, ln2b + (long)l * C, h);
    // ff = gelu(h @ W1 + b1)
    wgemm<<<g_ff1, 256, 0, stream>>>(h, C, w1T_l, C, FF, b1 + (long)l * FF, nullptr, ff, FF, 1);
    // x = x + ff @ W2 + b2
    wgemm<<<g_cc, 256, 0, stream>>>(ff, FF, w2T_l, FF, C, b2 + (long)l * C, x, x, C, 0);
  }

  // xout = lnf(x); logits = xout @ head_w^T
  ln_rows<<<BT, 256, 0, stream>>>(x, lnfw, lnfb, xout);
  dim3 g_head(V / 64, BT / 128);
  wgemm<<<g_head, 256, 0, stream>>>(xout, C, hdT, C, V, nullptr, nullptr, logits, V, 0);
}
